// LocalAggregation_75024488726595
// MI455X (gfx1250) — matrix operand order repaired and checked by reading, not compiled, not run on hardware
//
#include <hip/hip_runtime.h>

typedef __attribute__((ext_vector_type(16))) _Float16 v16h;
typedef __attribute__((ext_vector_type(8)))  _Float16 v8h;
typedef __attribute__((ext_vector_type(8)))  float    v8f;

namespace {
constexpr int   kN    = 50000;
constexpr int   kK    = 32;
constexpr int   kRows = kN * kK;      // 1,600,000 rows, divisible by 16
constexpr int   kC0   = 67;           // 64 features + 3 rel-coords (remapped)
constexpr int   kMid  = 64;
constexpr int   kOut  = 128;
constexpr int   kRBS  = 112;          // row stride in halfs: 224B, 32B aligned
constexpr int   kTPB  = 256;          // 8 wave32
constexpr int   kWPB  = 8;
constexpr float kEps  = 1e-5f;
}

// --------------------------------------------------------------------------
// Weight pre-swizzle into A-fragment order (Wt is the A operand).
// --------------------------------------------------------------------------
__device__ __forceinline__ void fill_w1frag(_Float16* w1f, const float* W1, int tid) {
  for (int i = tid; i < 12 * 32 * 16; i += kTPB) {   // s<3, t<4
    const int e   = i & 15;
    const int ln  = (i >> 4) & 31;
    const int st  = i >> 9;
    const int s   = st >> 2, t = st & 3;
    const int mid = 16 * t + (ln & 15);
    const int hi8 = (ln & 16) ? 8 : 0;
    const int k   = 32 * s + 2 * hi8 + e;
    float v = 0.f;
    if (k < 64)        v = W1[(3 + k) * kMid + mid];   // x-feature channel
    else if (k < kC0)  v = W1[(k - 64) * kMid + mid];  // dp channel
    w1f[i] = (_Float16)v;
  }
}

__device__ __forceinline__ void fill_w2frag(_Float16* w2f, const float* W2, int tid) {
  for (int i = tid; i < 16 * 32 * 16; i += kTPB) {   // s<2, t<8
    const int e   = i & 15;
    const int ln  = (i >> 4) & 31;
    const int st  = i >> 9;
    const int s   = st >> 3, t = st & 7;
    const int out = 16 * t + (ln & 15);
    const int hi8 = (ln & 16) ? 8 : 0;
    const int k   = 32 * s + 2 * hi8 + e;
    w2f[i] = (_Float16)W2[k * kOut + out];
  }
}

// --------------------------------------------------------------------------
// Row gather into a (double-buffered) LDS tile.
// AX=true: async DMA (global_load_async_to_lds_b128, ASYNCcnt) of the
// pre-converted f16 feature table; AX=false: f32 loads + cvt.
// Channels: [x0..x63 | dp0..dp2 | zeros]; pad pre-zeroed once.
// --------------------------------------------------------------------------
__device__ __forceinline__ void zero_rowbuf(_Float16* rb, int lane) {
  v8h z;
#pragma unroll
  for (int e = 0; e < 8; ++e) z[e] = (_Float16)0.f;
  for (int i = lane; i < 2 * 16 * kRBS / 8; i += 32) *(v8h*)(rb + 8 * i) = z;
}

template <bool AX>
__device__ __forceinline__ void gather_tile(_Float16* rb, const float* p, const float* x,
                                            const _Float16* x16, const int* idx,
                                            int rowbase, int lane) {
  const int r    = lane & 15;
  const int half = lane >> 4;
  const int row  = rowbase + r;
  const int j    = idx[row];
  if constexpr (AX) {
    // low 32 bits of a flat pointer into LDS == LDS byte address (ISA 10.2)
    const unsigned lds_off = (unsigned)(size_t)(rb + r * kRBS + half * 32);
    const unsigned goff    = (unsigned)(j * kMid + half * 32) * 2u;
    asm volatile("s_wait_dscnt 0" ::: "memory");     // buffer's prior reads done
    asm volatile(
        "global_load_async_to_lds_b128 %0, %1, %2 offset:0\n\t"
        "global_load_async_to_lds_b128 %0, %1, %2 offset:16\n\t"
        "global_load_async_to_lds_b128 %0, %1, %2 offset:32\n\t"
        "global_load_async_to_lds_b128 %0, %1, %2 offset:48"
        :: "v"(lds_off), "v"(goff), "s"(x16) : "memory");
  } else {
    const float4* xj = (const float4*)(x + 64 * j) + half * 8;
    _Float16* dst = rb + r * kRBS + half * 32;
#pragma unroll
    for (int q = 0; q < 4; ++q) {
      const float4 a = xj[2 * q], b = xj[2 * q + 1];
      v8h v;
      v[0] = (_Float16)a.x; v[1] = (_Float16)a.y; v[2] = (_Float16)a.z; v[3] = (_Float16)a.w;
      v[4] = (_Float16)b.x; v[5] = (_Float16)b.y; v[6] = (_Float16)b.z; v[7] = (_Float16)b.w;
      *(v8h*)(dst + 8 * q) = v;
    }
  }
  if (half) {                                        // overlaps the async DMA
    const int n = row / kK;
    const float* pj = p + 3 * j;
    const float* pn = p + 3 * n;
    _Float16* d2 = rb + r * kRBS;
    d2[64] = (_Float16)(pj[0] - pn[0]);
    d2[65] = (_Float16)(pj[1] - pn[1]);
    d2[66] = (_Float16)(pj[2] - pn[2]);
  }
}

template <bool AX>
__device__ __forceinline__ void wait_gather(bool next_inflight) {
  if constexpr (AX) {
    if (next_inflight) asm volatile("s_wait_asynccnt 0x4" ::: "memory");
    else               asm volatile("s_wait_asynccnt 0x0" ::: "memory");
  }
}

// --------------------------------------------------------------------------
// Transposed GEMMs: D = Wt x dataT; data is the B operand (K-contiguous 32B
// block per lane). All fragments loaded first, then the WMMA burst.
// C: M(channel) = 16t + (lane>=16?8:0)+e, N(row)=lane%16.
// --------------------------------------------------------------------------
__device__ __forceinline__ void gemm1_t(const _Float16* rb, const _Float16* w1f,
                                        int lane, v8f c[4]) {
  const int row = lane & 15;
  const int klo = (lane & 16) ? 16 : 0;
  v16h a[12], b[3];
#pragma unroll
  for (int s = 0; s < 3; ++s) {
    b[s] = *(const v16h*)(rb + row * kRBS + 32 * s + klo);
#pragma unroll
    for (int t = 0; t < 4; ++t)
      a[s * 4 + t] = *(const v16h*)(w1f + (((s * 4 + t) * 32 + lane) << 4));
  }
#pragma unroll
  for (int t = 0; t < 4; ++t)
#pragma unroll
    for (int e = 0; e < 8; ++e) c[t][e] = 0.f;
#pragma unroll
  for (int s = 0; s < 3; ++s)
#pragma unroll
    for (int t = 0; t < 4; ++t)
      c[t] = __builtin_amdgcn_wmma_f32_16x16x32_f16(false, a[s * 4 + t], false, b[s],
                                                    (short)0, c[t], false, false);
}

__device__ __forceinline__ void gemm2_t(const _Float16* h1, const _Float16* w2f,
                                        int lane, v8f c[8]) {
  const int row = lane & 15;
  const int klo = (lane & 16) ? 16 : 0;
  v16h a[16], b[2];
#pragma unroll
  for (int s = 0; s < 2; ++s) {
    b[s] = *(const v16h*)(h1 + row * kMid + 32 * s + klo);
#pragma unroll
    for (int t = 0; t < 8; ++t)
      a[s * 8 + t] = *(const v16h*)(w2f + (((s * 8 + t) * 32 + lane) << 4));
  }
#pragma unroll
  for (int t = 0; t < 8; ++t)
#pragma unroll
    for (int e = 0; e < 8; ++e) c[t][e] = 0.f;
#pragma unroll
  for (int s = 0; s < 2; ++s)
#pragma unroll
    for (int t = 0; t < 8; ++t)
      c[t] = __builtin_amdgcn_wmma_f32_16x16x32_f16(false, a[s * 8 + t], false, b[s],
                                                    (short)0, c[t], false, false);
}

template <int NT>
__device__ __forceinline__ void load_affine(const float* scg, const float* shg,
                                            v8f* sc, v8f* sh, int lane) {
  const int mb = (lane & 16) ? 8 : 0;
#pragma unroll
  for (int t = 0; t < NT; ++t)
#pragma unroll
    for (int e = 0; e < 8; ++e) {
      sc[t][e] = scg[16 * t + mb + e];
      sh[t][e] = shg[16 * t + mb + e];
    }
}

__device__ __forceinline__ void norm_store_h1(const v8f c[4], const v8f sc[4],
                                              const v8f sh[4], _Float16* h1, int lane) {
  const int row = lane & 15;
  const int mb  = (lane & 16) ? 8 : 0;
#pragma unroll
  for (int t = 0; t < 4; ++t) {
    v8h pk;
#pragma unroll
    for (int e = 0; e < 8; ++e)
      pk[e] = (_Float16)fmaxf(fmaf(c[t][e], sc[t][e], sh[t][e]), 0.f);
    *(v8h*)(h1 + row * kMid + 16 * t + mb) = pk;
  }
}

template <int NT>
__device__ __forceinline__ void fold_stats(v8f* s, v8f* q, float* accw, int lane) {
#pragma unroll
  for (int t = 0; t < NT; ++t) {
#pragma unroll
    for (int st = 1; st < 16; st <<= 1)
#pragma unroll
      for (int e = 0; e < 8; ++e) {
        s[t][e] += __shfl_xor(s[t][e], st);
        q[t][e] += __shfl_xor(q[t][e], st);
      }
    if ((lane & 15) == 0) {
      const int mb = (lane & 16) ? 8 : 0;
#pragma unroll
      for (int e = 0; e < 8; ++e) {
        accw[16 * t + mb + e]           = s[t][e];
        accw[NT * 16 + 16 * t + mb + e] = q[t][e];
      }
    }
  }
}

// ---------------- kernels ---------------------------------------------------

__global__ void la_cvt_x(const float* __restrict__ x, _Float16* __restrict__ x16) {
  const int i = blockIdx.x * kTPB + threadIdx.x;     // 8 values per thread
  if (i < kN * kMid / 8) {
    const float4 a = ((const float4*)x)[2 * i];
    const float4 b = ((const float4*)x)[2 * i + 1];
    v8h v;
    v[0] = (_Float16)a.x; v[1] = (_Float16)a.y; v[2] = (_Float16)a.z; v[3] = (_Float16)a.w;
    v[4] = (_Float16)b.x; v[5] = (_Float16)b.y; v[6] = (_Float16)b.z; v[7] = (_Float16)b.w;
    ((v8h*)x16)[i] = v;
  }
}

template <bool AX>
__global__ __launch_bounds__(kTPB) void la_stats1(const float* __restrict__ p,
                                                  const float* __restrict__ x,
                                                  const _Float16* __restrict__ x16,
                                                  const int* __restrict__ idx,
                                                  const float* __restrict__ W1,
                                                  float* __restrict__ stat) {
  __shared__ __align__(32) _Float16 w1f[12 * 32 * 16];
  __shared__ __align__(32) _Float16 rowbuf[kWPB][2][16 * kRBS];
  __shared__ float acc[kWPB][128];
  const int tid = threadIdx.x, wave = tid >> 5, lane = tid & 31;

  fill_w1frag(w1f, W1, tid);
  zero_rowbuf(&rowbuf[wave][0][0], lane);
  __syncthreads();

  v8f s[4], q[4];
#pragma unroll
  for (int t = 0; t < 4; ++t)
#pragma unroll
    for (int e = 0; e < 8; ++e) { s[t][e] = 0.f; q[t][e] = 0.f; }

  const int gwave  = blockIdx.x * kWPB + wave;
  const int nwaves = gridDim.x * kWPB;
  const int ntiles = kRows / 16;
  int ib = 0;
  if (gwave < ntiles)
    gather_tile<AX>(&rowbuf[wave][0][0], p, x, x16, idx, gwave * 16, lane);
  for (int tile = gwave; tile < ntiles; tile += nwaves) {
    const int nxt = tile + nwaves;
    const bool has_next = nxt < ntiles;
    if (has_next)
      gather_tile<AX>(&rowbuf[wave][ib ^ 1][0], p, x, x16, idx, nxt * 16, lane);
    wait_gather<AX>(has_next);
    v8f c[4];
    gemm1_t(&rowbuf[wave][ib][0], w1f, lane, c);
#pragma unroll
    for (int t = 0; t < 4; ++t)
#pragma unroll
      for (int e = 0; e < 8; ++e) {
        const float v = c[t][e];
        s[t][e] += v;
        q[t][e]  = fmaf(v, v, q[t][e]);
      }
    ib ^= 1;
  }
  fold_stats<4>(s, q, &acc[wave][0], lane);
  __syncthreads();
  for (int ch = tid; ch < 128; ch += kTPB) {
    float sum = 0.f;
#pragma unroll
    for (int w = 0; w < kWPB; ++w) sum += acc[w][ch];
    stat[blockIdx.x * 128 + ch] = sum;
  }
}

__global__ void la_reduce1(const float* __restrict__ g1, const float* __restrict__ bt1,
                           float* __restrict__ stat, int sb) {
  const int c = threadIdx.x;                         // 64 threads
  float s = 0.f, q = 0.f;
  for (int g = 0; g < sb; ++g) { s += stat[g * 128 + c]; q += stat[g * 128 + 64 + c]; }
  const float inv  = 1.0f / (float)kRows;
  const float mean = s * inv;
  const float var  = fmaxf(q * inv - mean * mean, 0.f);
  const float sc   = g1[c] * rsqrtf(var + kEps);
  float* sc1 = stat + sb * 384;
  sc1[c]      = sc;
  sc1[64 + c] = bt1[c] - mean * sc;                  // bias cancels in BN
}

template <bool AX>
__global__ __launch_bounds__(kTPB) void la_stats2(const float* __restrict__ p,
                                                  const float* __restrict__ x,
                                                  const _Float16* __restrict__ x16,
                                                  const int* __restrict__ idx,
                                                  const float* __restrict__ W1,
                                                  const float* __restrict__ W2,
                                                  float* __restrict__ stat, int sb) {
  __shared__ __align__(32) _Float16 w1f[12 * 32 * 16];
  __shared__ __align__(32) _Float16 w2f[16 * 32 * 16];
  __shared__ __align__(32) _Float16 rowbuf[kWPB][2][16 * kRBS];
  __shared__ __align__(32) _Float16 h1buf[kWPB][16 * kMid];
  __shared__ float acc[kWPB][256];
  const int tid = threadIdx.x, wave = tid >> 5, lane = tid & 31;

  fill_w1frag(w1f, W1, tid);
  fill_w2frag(w2f, W2, tid);
  zero_rowbuf(&rowbuf[wave][0][0], lane);
  __syncthreads();

  v8f sc1r[4], sh1r[4];
  load_affine<4>(stat + sb * 384, stat + sb * 384 + 64, sc1r, sh1r, lane);

  v8f s[8], q[8];
#pragma unroll
  for (int t = 0; t < 8; ++t)
#pragma unroll
    for (int e = 0; e < 8; ++e) { s[t][e] = 0.f; q[t][e] = 0.f; }

  const int gwave  = blockIdx.x * kWPB + wave;
  const int nwaves = gridDim.x * kWPB;
  const int ntiles = kRows / 16;
  int ib = 0;
  if (gwave < ntiles)
    gather_tile<AX>(&rowbuf[wave][0][0], p, x, x16, idx, gwave * 16, lane);
  for (int tile = gwave; tile < ntiles; tile += nwaves) {
    const int nxt = tile + nwaves;
    const bool has_next = nxt < ntiles;
    if (has_next)
      gather_tile<AX>(&rowbuf[wave][ib ^ 1][0], p, x, x16, idx, nxt * 16, lane);
    wait_gather<AX>(has_next);
    v8f c1[4];
    gemm1_t(&rowbuf[wave][ib][0], w1f, lane, c1);
    norm_store_h1(c1, sc1r, sh1r, &h1buf[wave][0], lane);
    v8f c2[8];
    gemm2_t(&h1buf[wave][0], w2f, lane, c2);
#pragma unroll
    for (int t = 0; t < 8; ++t)
#pragma unroll
      for (int e = 0; e < 8; ++e) {
        const float v = c2[t][e];
        s[t][e] += v;
        q[t][e]  = fmaf(v, v, q[t][e]);
      }
    ib ^= 1;
  }
  fold_stats<8>(s, q, &acc[wave][0], lane);
  __syncthreads();
  float* P2 = stat + sb * 128;
  for (int ch = tid; ch < 256; ch += kTPB) {
    float sum = 0.f;
#pragma unroll
    for (int w = 0; w < kWPB; ++w) sum += acc[w][ch];
    P2[blockIdx.x * 256 + ch] = sum;
  }
}

__global__ void la_reduce2(const float* __restrict__ g2, const float* __restrict__ bt2,
                           float* __restrict__ stat, int sb) {
  const int c = threadIdx.x;                         // 128 threads
  const float* P2 = stat + sb * 128;
  float s = 0.f, q = 0.f;
  for (int g = 0; g < sb; ++g) { s += P2[g * 256 + c]; q += P2[g * 256 + 128 + c]; }
  const float inv  = 1.0f / (float)kRows;
  const float mean = s * inv;
  const float var  = fmaxf(q * inv - mean * mean, 0.f);
  const float sc   = g2[c] * rsqrtf(var + kEps);
  float* sc2 = stat + sb * 384 + 128;
  sc2[c]       = sc;
  sc2[128 + c] = bt2[c] - mean * sc;
}

template <bool AX>
__global__ __launch_bounds__(kTPB) void la_final(const float* __restrict__ p,
                                                 const float* __restrict__ x,
                                                 const _Float16* __restrict__ x16,
                                                 const int* __restrict__ idx,
                                                 const float* __restrict__ W1,
                                                 const float* __restrict__ W2,
                                                 const float* __restrict__ stat, int sb,
                                                 float* __restrict__ out) {
  __shared__ __align__(32) _Float16 w1f[12 * 32 * 16];
  __shared__ __align__(32) _Float16 w2f[16 * 32 * 16];
  __shared__ __align__(32) _Float16 rowbuf[kWPB][2][16 * kRBS];
  __shared__ __align__(32) _Float16 h1buf[kWPB][16 * kMid];
  const int tid = threadIdx.x, wave = tid >> 5, lane = tid & 31;

  fill_w1frag(w1f, W1, tid);
  fill_w2frag(w2f, W2, tid);
  zero_rowbuf(&rowbuf[wave][0][0], lane);
  __syncthreads();

  const float* base = stat + sb * 384;
  v8f sc1r[4], sh1r[4], sc2r[8], sh2r[8];
  load_affine<4>(base, base + 64, sc1r, sh1r, lane);
  load_affine<8>(base + 128, base + 256, sc2r, sh2r, lane);

  const int n = blockIdx.x * kWPB + wave;            // one wave per point
  v8f mx[8];
#pragma unroll
  for (int t = 0; t < 8; ++t)
#pragma unroll
    for (int e = 0; e < 8; ++e) mx[t][e] = 0.f;      // ReLU output >= 0

  gather_tile<AX>(&rowbuf[wave][0][0], p, x, x16, idx, n * kK, lane);
  gather_tile<AX>(&rowbuf[wave][1][0], p, x, x16, idx, n * kK + 16, lane);
#pragma unroll
  for (int half = 0; half < 2; ++half) {             // K=32 -> two 16-row tiles
    wait_gather<AX>(half == 0);                      // <=4 then 0: tile data ready
    v8f c1[4];
    gemm1_t(&rowbuf[wave][half][0], w1f, lane, c1);
    norm_store_h1(c1, sc1r, sh1r, &h1buf[wave][0], lane);
    v8f c2[8];
    gemm2_t(&h1buf[wave][0], w2f, lane, c2);
#pragma unroll
    for (int t = 0; t < 8; ++t)
#pragma unroll
      for (int e = 0; e < 8; ++e)
        mx[t][e] = fmaxf(mx[t][e],
                         fmaxf(fmaf(c2[t][e], sc2r[t][e], sh2r[t][e]), 0.f));
  }
#pragma unroll
  for (int st = 1; st < 16; st <<= 1)
#pragma unroll
    for (int t = 0; t < 8; ++t)
#pragma unroll
      for (int e = 0; e < 8; ++e)
        mx[t][e] = fmaxf(mx[t][e], __shfl_xor(mx[t][e], st));

  const int l4 = lane & 15;
  const int tw = l4 >> 1;
  const int e0 = (lane & 1) * 4;
  const int mb = (lane & 16) ? 8 : 0;
  v8f sel = mx[0];
#pragma unroll
  for (int tt = 1; tt < 8; ++tt) if (tw == tt) sel = mx[tt];
  float4 o;
  if (e0 == 0) o = make_float4(sel[0], sel[1], sel[2], sel[3]);
  else         o = make_float4(sel[4], sel[5], sel[6], sel[7]);
  *(float4*)(out + n * kOut + 16 * tw + mb + e0) = o;
}

// ---------------- launcher --------------------------------------------------

extern "C" void kernel_launch(void* const* d_in, const int* in_sizes, int n_in,
                              void* d_out, int out_size, void* d_ws, size_t ws_size,
                              hipStream_t stream) {
  (void)in_sizes; (void)n_in; (void)out_size;
  const float* p   = (const float*)d_in[0];
  const float* x   = (const float*)d_in[1];
  const int*   idx = (const int*)  d_in[2];
  const float* W1  = (const float*)d_in[3];
  const float* g1  = (const float*)d_in[5];
  const float* bt1 = (const float*)d_in[6];
  const float* W2  = (const float*)d_in[7];
  const float* g2  = (const float*)d_in[9];
  const float* bt2 = (const float*)d_in[10];
  float* out = (float*)d_out;
  float* ws  = (float*)d_ws;

  const long xoffF = (long)kN * kMid / 2;            // f16 feature table, in floats
  const long wsf   = (long)(ws_size / sizeof(float));

  bool ax = false;
  int  sb;
  const long rem16 = wsf - xoffF - 384;
  if (rem16 / 384 >= 16) {                           // room for x16 + >=16 partials
    ax = true;
    sb = (int)(rem16 / 384 > 512 ? 512 : rem16 / 384);
  } else {
    long r = (wsf - 384) / 384;
    sb = (int)(r > 512 ? 512 : (r < 1 ? 1 : r));
  }

  _Float16* x16 = (_Float16*)ws;
  float*    st  = ax ? (ws + xoffF) : ws;

  if (ax) {
    const int nconv = kN * kMid / 8;
    la_cvt_x<<<(nconv + kTPB - 1) / kTPB, kTPB, 0, stream>>>(x, x16);
    la_stats1<true><<<sb, kTPB, 0, stream>>>(p, x, x16, idx, W1, st);
    la_reduce1<<<1, 64, 0, stream>>>(g1, bt1, st, sb);
    la_stats2<true><<<sb, kTPB, 0, stream>>>(p, x, x16, idx, W1, W2, st, sb);
    la_reduce2<<<1, 128, 0, stream>>>(g2, bt2, st, sb);
    la_final<true><<<kN / kWPB, kTPB, 0, stream>>>(p, x, x16, idx, W1, W2, st, sb, out);
  } else {
    la_stats1<false><<<sb, kTPB, 0, stream>>>(p, x, x16, idx, W1, st);
    la_reduce1<<<1, 64, 0, stream>>>(g1, bt1, st, sb);
    la_stats2<false><<<sb, kTPB, 0, stream>>>(p, x, x16, idx, W1, W2, st, sb);
    la_reduce2<<<1, 128, 0, stream>>>(g2, bt2, st, sb);
    la_final<false><<<kN / kWPB, kTPB, 0, stream>>>(p, x, x16, idx, W1, W2, st, sb, out);
  }
}
